// FirConv_90778428769019
// MI455X (gfx1250) — matrix operand order repaired and checked by reading, not compiled, not run on hardware
//
#include <hip/hip_runtime.h>
#include <hip/hip_bf16.h>
#include <math.h>

// ---------------------------------------------------------------------------
// FIR zero-phase filter on MI455X (gfx1250).
// f16 WMMA (f32 accumulate) banded-Toeplitz GEMM + async global->LDS staging.
// x: [64, 64, 10000] f32, weight: [100] f32  ->  out: [64, 64, 10000] f32
// ---------------------------------------------------------------------------

typedef _Float16 v4h  __attribute__((ext_vector_type(4)));
typedef _Float16 v8h  __attribute__((ext_vector_type(8)));
typedef _Float16 v16h __attribute__((ext_vector_type(16)));
typedef float    v8f  __attribute__((ext_vector_type(8)));
typedef float    v4f  __attribute__((ext_vector_type(4)));
typedef int      b64v __attribute__((vector_size(2 * sizeof(int))));  // async b64 payload

#define AS1 __attribute__((address_space(1)))
#define AS3 __attribute__((address_space(3)))

#define TLEN   10000        // samples per row
#define NCH    4096         // 64*64 rows
#define KT     1001         // FIR taps
#define PAD    500          // (KT-1)/2

// Padded f16 row layout in workspace: zeros implement 'same' zero-padding and
// make every staging copy alignment-friendly and guard-free.
#define XP     11136        // padded row pitch (halfs), multiple of 8
#define DOFF   516          // data offset within padded row (DOFF-PAD = 16)

// Per-block tiling: 256 threads = 8 wave32, each wave owns a 16(time)x16(ch) tile.
#define WAVES      8
#define TPB        (WAVES * 32)
#define TGROUP     (WAVES * 16)            // 128 time samples per block
#define XCOLS      1136                    // window: 16*7 + 32*31 + 31 + pad
#define XPITCH     1144                    // LDS pitch (halfs), mult. of 8, bank-skewed
#define ALEN       1040                    // k[d] for d in [-16, 1024)

// --- CDNA5 async global->LDS path (hedged: falls back to plain LDS copies) --
#if defined(__has_builtin)
#  if __has_builtin(__builtin_amdgcn_global_load_async_to_lds_b64)
#    define USE_ASYNC_LDS 1
#  endif
#endif
#ifndef USE_ASYNC_LDS
#  define USE_ASYNC_LDS 0
#endif

__device__ __forceinline__ void wait_async_lds() {
#if defined(__has_builtin)
#  if __has_builtin(__builtin_amdgcn_s_wait_asynccnt)
  __builtin_amdgcn_s_wait_asynccnt(0);
  return;
#  endif
#endif
  asm volatile("s_wait_asynccnt 0x0" ::: "memory");
}

__device__ __forceinline__ float dev_sinc(float x) {
  const float px = 3.14159265358979323846f * x;
  return (x == 0.0f) ? 1.0f : (sinf(px) / px);
}

// ---------------------------------------------------------------------------
// Kernel 1: h[j] = sum_f [w_hard(f) * (r*sinc(r*m) - l*sinc(l*m))] * hamming(j)
// Forward of the straight-through estimator is exactly w_hard = (weight >= 0).
// ---------------------------------------------------------------------------
__global__ void fir_build_kernel(const float* __restrict__ weight,
                                 float* __restrict__ kern) {
  const int j = blockIdx.x * blockDim.x + threadIdx.x;
  if (j >= KT) return;
  const float m = (float)j - (float)PAD;
  float acc = 0.0f;
  for (int f = 0; f < 100; ++f) {
    if (weight[f] >= 0.0f) {   // sigmoid(w) >= 0.5  <=>  w >= 0
      const float l = (float)(f + 1) * (1.0f / 125.0f);
      const float r = (float)(f + 2) * (1.0f / 125.0f);
      acc += r * dev_sinc(r * m) - l * dev_sinc(l * m);
    }
  }
  // torch.hamming_window periodic: denominator N (= 1001)
  const float ham =
      0.54f - 0.46f * cosf(6.283185307179586f * (float)j / (float)KT);
  kern[j] = acc * ham;
}

// ---------------------------------------------------------------------------
// Kernel 2: build padded f16 image. x != null: convert+pad x. x == null: zeros
// (pre-clears the intermediate's pad region; its data region is overwritten).
// ---------------------------------------------------------------------------
__global__ __launch_bounds__(256) void convert_pad(const float* __restrict__ x,
                                                   _Float16* __restrict__ dst) {
  const size_t v = (size_t)blockIdx.x * blockDim.x + threadIdx.x;  // vec8 id
  if (v >= (size_t)NCH * (XP / 8)) return;
  const int ch = (int)(v / (XP / 8));
  const int c8 = (int)(v % (XP / 8)) * 8;
  v8h o;
  if (x && c8 >= DOFF && c8 + 8 <= DOFF + TLEN) {          // interior: vector
    const float* xp = x + (size_t)ch * TLEN + (c8 - DOFF); // 16B-aligned
    union { v4f q; float f[4]; } a, b;
    a.q = *(const v4f*)xp;
    b.q = *(const v4f*)(xp + 4);
#pragma unroll
    for (int i = 0; i < 4; ++i) { o[i] = (_Float16)a.f[i]; o[4 + i] = (_Float16)b.f[i]; }
  } else {                                                 // pad / boundary
#pragma unroll
    for (int i = 0; i < 8; ++i) {
      const int c = c8 + i - DOFF;
      const float val = (x && c >= 0 && c < TLEN) ? x[(size_t)ch * TLEN + c] : 0.0f;
      o[i] = (_Float16)val;
    }
  }
  *(v8h*)(dst + (size_t)ch * XP + c8) = o;
}

// ---------------------------------------------------------------------------
// Kernel 3/4: one correlation pass as banded-Toeplitz GEMM via WMMA.
//   D[m][n] = sum_q A[m][q] * X[n][p0+q],  A[m][q] = k[q - m]  (band, else 0)
// A fragments come from a single shifted 1-D kernel line in LDS (2 KB).
// X window staged via GLOBAL_LOAD_ASYNC_TO_LDS_B64 (ASYNCcnt), guard-free
// thanks to the padded layout. reverse=1 uses k[1000-d].
// ---------------------------------------------------------------------------
__global__ __launch_bounds__(TPB) void fir_pass(
    const _Float16* __restrict__ src,      // padded f16 [NCH][XP]
    float*          __restrict__ dstF32,   // pass 2 dest (or null)
    _Float16*       __restrict__ dstF16,   // pass 1 dest, padded (or null)
    const float*    __restrict__ kern,
    const int reverse) {
  __shared__ _Float16 Ak[ALEN];            // Ak[d + 16] = k[d], zero outside band
  __shared__ _Float16 Xs[16 * XPITCH];     // 16 channels x staged time window

  const int tid = threadIdx.x;
  const int g   = blockIdx.x * TGROUP;     // time-group base
  const int c0  = blockIdx.y * 16;         // channel-group base

  // Stage (possibly reversed) kernel line, f16.
  for (int i = tid; i < ALEN; i += TPB) {
    const int d = i - 16;
    float v = 0.0f;
    if (d >= 0 && d < KT) v = kern[reverse ? (KT - 1 - d) : d];
    Ak[i] = (_Float16)v;
  }

  // Stage 16ch x XCOLS f16 window. Padded-row col base g+16 is 16B-aligned,
  // window never crosses a padded row -> no bounds checks anywhere.
  {
    const int gc0 = g + (DOFF - PAD);      // = g + 16
    for (int e = tid; e < 16 * (XCOLS / 4); e += TPB) {
      const int ch  = e / (XCOLS / 4);
      const int col = (e - ch * (XCOLS / 4)) * 4;          // 8B-aligned
      const _Float16* gp = src + (size_t)(c0 + ch) * XP + gc0 + col;
#if USE_ASYNC_LDS
      __builtin_amdgcn_global_load_async_to_lds_b64(
          (AS1 b64v*)gp, (AS3 b64v*)&Xs[ch * XPITCH + col], 0, 0);
#else
      *(v4h*)&Xs[ch * XPITCH + col] = *(const v4h*)gp;
#endif
    }
#if USE_ASYNC_LDS
    wait_async_lds();
#endif
  }
  __syncthreads();

  const int wave = tid >> 5;
  const int lane = tid & 31;
  const int t0   = g + wave * 16;
  if (t0 >= TLEN) return;                  // wave-uniform; EXEC stays all-1s

  const int row   = lane & 15;             // A: M row / B: N column / D: N column
  const int hi    = lane >> 4;             // half-wave selector
  const int aoff0 = hi ? 8 : 0;            // 16-bit A 16x32 layout: K blocks
  const int aoff1 = hi ? 24 : 16;
  const int abase = 16 - row;              // Ak index for d = q - row
  const _Float16* __restrict__ xrow = &Xs[row * XPITCH + wave * 16];

  union VA { v16h v; v8h h[2]; _Float16 e[16]; };
  v8f c = {};

  for (int s = 0; s < 32; ++s) {           // K = 1024 in 32 x k32 steps
    const int q0 = s * 32;
    VA a, b;
#pragma unroll
    for (int i = 0; i < 8; ++i) {          // banded-Toeplitz A frag
      a.e[i]     = Ak[abase + q0 + aoff0 + i];
      a.e[8 + i] = Ak[abase + q0 + aoff1 + i];
    }
    const int bq = q0 + 8 * hi;            // 16-bit B 32x16 layout: K blocks 8h, 16+8h, 16B-aligned
    b.h[0] = *(const v8h*)(xrow + bq);
    b.h[1] = *(const v8h*)(xrow + bq + 16);
    c = __builtin_amdgcn_wmma_f32_16x16x32_f16(
        /*neg_a=*/false, a.v, /*neg_b=*/false, b.v,
        /*c_mod=*/(short)0, c, /*reuse_a=*/false, /*reuse_b=*/false);
  }

  // D layout: lane -> channel c0+row; VGPR r -> time t0 + 8*hi + r (contiguous).
  const int ch = c0 + row;
  const int tw = t0 + hi * 8;
  if (dstF32) {                            // pass 2: unpadded f32 out, 16B stores
    const size_t off = (size_t)ch * TLEN + tw;
    union { v8f v; v4f q[2]; } u; u.v = c;
    *(v4f*)(dstF32 + off)     = u.q[0];
    *(v4f*)(dstF32 + off + 4) = u.q[1];
  } else {                                 // pass 1: padded f16, 8B-aligned stores
    v4h h0, h1;
#pragma unroll
    for (int i = 0; i < 4; ++i) { h0[i] = (_Float16)c[i]; h1[i] = (_Float16)c[4 + i]; }
    const size_t off = (size_t)ch * XP + DOFF + tw;
    *(v4h*)(dstF16 + off)     = h0;
    *(v4h*)(dstF16 + off + 4) = h1;
  }
}

// ---------------------------------------------------------------------------
// ws layout: [0,4KB) f32 taps; then xpad (91.2MB); then y1pad (91.2MB).
// Both intermediates are L2-resident (192MB global L2).
// ---------------------------------------------------------------------------
extern "C" void kernel_launch(void* const* d_in, const int* in_sizes, int n_in,
                              void* d_out, int out_size, void* d_ws, size_t ws_size,
                              hipStream_t stream) {
  const float* x      = (const float*)d_in[0];
  const float* weight = (const float*)d_in[1];
  float*     kern  = (float*)d_ws;
  _Float16*  xpad  = (_Float16*)((char*)d_ws + 4096);
  _Float16*  y1pad = xpad + (size_t)NCH * XP;
  float*     out   = (float*)d_out;

  fir_build_kernel<<<dim3((KT + 127) / 128), dim3(128), 0, stream>>>(weight, kern);

  const size_t nv8 = (size_t)NCH * (XP / 8);
  const int cblocks = (int)((nv8 + 255) / 256);
  convert_pad<<<dim3(cblocks), dim3(256), 0, stream>>>(x, xpad);        // f32 -> padded f16
  convert_pad<<<dim3(cblocks), dim3(256), 0, stream>>>(nullptr, y1pad); // zero pad regions

  dim3 grid((TLEN + TGROUP - 1) / TGROUP, NCH / 16);   // (79, 256)
  // Pass 1: correlation with k, xpad -> y1pad (f16)
  fir_pass<<<grid, dim3(TPB), 0, stream>>>(xpad, nullptr, y1pad, kern, 0);
  // Pass 2: correlation with reversed k, y1pad -> out (f32)
  fir_pass<<<grid, dim3(TPB), 0, stream>>>(y1pad, out, nullptr, kern, 1);
}
